// MoELayer_16166256902775
// MI455X (gfx1250) — matrix operand order repaired and checked by reading, not compiled, not run on hardware
//
#include <hip/hip_runtime.h>
#include <hip/hip_bf16.h>

// ---------------------------------------------------------------------------
// MoE top-2 dispatch for MI455X (gfx1250), wave32 + WMMA bf16.
//   B=2, T=2048 -> N=4096 tokens, D=1024, F=4096, E=8, TOP_K=2
// Sparse dispatch = 137 GFLOP bf16 (4x less than dense reference math).
// Weights pre-converted fp32 -> bf16 into workspace each call (+6us HBM);
// hot loop is pure bf16 b128 loads feeding v_wmma_f32_16x16x32_bf16.
// 32-token tiles: two 16-row A tiles share every B fragment -> 32 bf16
// FLOP per L2 byte; LDS = 320KB (full WGP), 1 block/WGP, 2 waves/SIMD.
// ---------------------------------------------------------------------------

typedef __bf16 v16bf __attribute__((ext_vector_type(16)));
typedef float  v8f   __attribute__((ext_vector_type(8)));

constexpr int D = 1024;
constexpr int F = 4096;
constexpr int E = 8;
constexpr int N = 4096;          // B*T tokens
constexpr int TILE = 32;         // tokens per expert workgroup (2 WMMA M-tiles)
constexpr int NTILES = N / TILE; // 128 (worst case: all tokens -> one expert)

// ---------------- kernel 0: zero output + per-expert counters --------------
__global__ void moe_zero_kernel(float* __restrict__ out, int* __restrict__ cnt,
                                int n4) {
  int i = blockIdx.x * blockDim.x + threadIdx.x;
  if (i < n4) ((float4*)out)[i] = make_float4(0.f, 0.f, 0.f, 0.f);
  if (i < E) cnt[i] = 0;
}

// ---------------- kernel 0b: fp32 -> bf16 weight convert -------------------
__global__ void moe_cvt_kernel(const float* __restrict__ src,
                               __bf16* __restrict__ dst, int n8) {
  int i = blockIdx.x * blockDim.x + threadIdx.x;
  if (i >= n8) return;
  const float4* s = (const float4*)src + 2 * (size_t)i;
  float4 a = s[0], b = s[1];
  alignas(16) __bf16 v[8] = {(__bf16)a.x, (__bf16)a.y, (__bf16)a.z, (__bf16)a.w,
                             (__bf16)b.x, (__bf16)b.y, (__bf16)b.z, (__bf16)b.w};
  *(uint4*)(dst + 8 * (size_t)i) = *(const uint4*)v;
}

// ---------------- kernel 1: router (logits -> top2 -> scatter) -------------
// 16 blocks x 256 threads; one thread per token; Wr (8x1024 fp32 = 32KB) in
// LDS. Renormalized top-2 softmax weights: g1 = 1/(1+exp(l0-l1)); the full
// softmax denominator cancels.
__global__ void moe_router_kernel(const float* __restrict__ x,
                                  const float* __restrict__ Wr,
                                  int* __restrict__ cnt,
                                  int* __restrict__ tokIdx,
                                  float* __restrict__ tokGate) {
  __shared__ float wr[E * D];
  const int tid = threadIdx.x;
  for (int i = tid; i < E * D; i += blockDim.x) wr[i] = Wr[i];
  __syncthreads();

  const int n = blockIdx.x * blockDim.x + tid;
  const float4* xr = (const float4*)(x + (size_t)n * D);

  float acc[E];
#pragma unroll
  for (int e = 0; e < E; ++e) acc[e] = 0.f;

  for (int d4 = 0; d4 < D / 4; ++d4) {
    float4 xv = xr[d4];
#pragma unroll
    for (int e = 0; e < E; ++e) {
      float4 wv = ((const float4*)(wr + e * D))[d4];
      acc[e] += xv.x * wv.x + xv.y * wv.y + xv.z * wv.z + xv.w * wv.w;
    }
  }

  // top-2 (strict > keeps lowest index on ties, matching lax.top_k)
  int e0 = 0; float l0 = acc[0];
#pragma unroll
  for (int e = 1; e < E; ++e) if (acc[e] > l0) { l0 = acc[e]; e0 = e; }
  int e1 = (e0 == 0) ? 1 : 0; float l1 = acc[e1];
#pragma unroll
  for (int e = 0; e < E; ++e)
    if (e != e0 && acc[e] > l1) { l1 = acc[e]; e1 = e; }

  float g1 = 1.f / (1.f + expf(l0 - l1));  // p1/(p0+p1)
  float g0 = 1.f - g1;

  int p0 = atomicAdd(&cnt[e0], 1);
  tokIdx[e0 * N + p0] = n;  tokGate[e0 * N + p0] = g0;
  int p1 = atomicAdd(&cnt[e1], 1);
  tokIdx[e1 * N + p1] = n;  tokGate[e1 * N + p1] = g1;
}

// ---------------- helpers ---------------------------------------------------
// B 32x16 (CDNA5 7.12.2): lane L -> col L&15; K-base 0 / 8 by lane half;
// elements 0..7 -> K=base+0..7, 8..15 -> K=base+16..23.
__device__ inline void loadB(const __bf16* p, v16bf& b) {
  ((uint4*)&b)[0] = *(const uint4*)p;
  ((uint4*)&b)[1] = *(const uint4*)(p + 16);
}
__device__ inline void loadB(const float* p, v16bf& b) {
#pragma unroll
  for (int i = 0; i < 16; ++i) b[i] = (__bf16)p[i < 8 ? i : i + 8];
}
// A 16x32 (CDNA5 7.12.2): lane L -> row L&15; K-base 0 / 8 by lane half;
// elements 0..7 -> K=base+0..7, 8..15 -> K=base+16..23.
__device__ inline void loadA(const __bf16* p, v16bf& a) {
  ((uint4*)&a)[0] = *(const uint4*)p;
  ((uint4*)&a)[1] = *(const uint4*)(p + 16);
}
__device__ inline float gelu_exact(float v) {
  return 0.5f * v * (1.f + erff(v * 0.70710678118654752f));
}
__device__ inline v8f wmma_bf16(const v16bf& a, const v16bf& b, const v8f& c) {
  return __builtin_amdgcn_wmma_f32_16x16x32_bf16(false, a, false, b, (short)0,
                                                 c, false, false);
}

// ---------------- kernel 2: fused expert FFN (gelu(X W1^T) W2^T) -----------
// Block = 256 threads = 8 waves; one (expert, 32-token tile) per block.
// LDS: Xs 32x1024 bf16 (64KB) + Hs 32x4096 bf16 (256KB) = 320KB (full WGP).
// Inner loop: 2 A frags + 2 B frags -> 4 WMMAs (each B reused by both M-tiles).
template <typename WT>
__global__ __launch_bounds__(256) void moe_expert_kernel(
    const float* __restrict__ x, const WT* __restrict__ W1,
    const WT* __restrict__ W2, const int* __restrict__ cnt,
    const int* __restrict__ tokIdx, const float* __restrict__ tokGate,
    float* __restrict__ out) {
  extern __shared__ char smem[];
  __bf16* Xs = (__bf16*)smem;                           // TILE * D   (64KB)
  __bf16* Hs = (__bf16*)(smem + (size_t)TILE * D * 2);  // TILE * F  (256KB)

  const int e = blockIdx.y;
  const int t = blockIdx.x;
  const int c = cnt[e];
  const int n0 = t * TILE;
  if (n0 >= c) return;

  const int tid   = threadIdx.x;
  const int lane  = tid & 31;
  const int wv    = tid >> 5;               // wave 0..7
  const int hrow  = lane & 15;              // A row / B col / C col
  const int aBase = (lane < 16) ? 0 : 8;    // A K-base
  const int bBase = (lane < 16) ? 0 : 8;    // B K-base
  const int crow  = (lane < 16) ? 0 : 8;    // C row offset

  // gather X tile -> bf16 LDS (each thread: one row, 128 contiguous floats)
  {
    const int row = tid >> 3;               // 0..31
    const int seg = tid & 7;                // 128 floats each
    const int tok = ((n0 + row) < c) ? tokIdx[e * N + n0 + row] : 0;
    const float4* src = (const float4*)(x + (size_t)tok * D) + seg * 32;
    __bf16* dst = Xs + row * D + seg * 128;
#pragma unroll
    for (int i = 0; i < 32; ++i) {
      float4 v = src[i];
      dst[i * 4 + 0] = (__bf16)v.x; dst[i * 4 + 1] = (__bf16)v.y;
      dst[i * 4 + 2] = (__bf16)v.z; dst[i * 4 + 3] = (__bf16)v.w;
    }
  }
  __syncthreads();

  const WT* W1e = W1 + (size_t)e * F * D;   // [F, D] row-major
  const WT* W2e = W2 + (size_t)e * D * F;   // [D, F] row-major

  // ---- phase 1: Hs = gelu(Xs @ W1e^T); wave wv owns f-tiles [32wv, 32wv+32)
  for (int ft = wv * 32; ft < wv * 32 + 32; ft += 2) {
    v8f acc00 = {}, acc01 = {}, acc10 = {}, acc11 = {};
    const int f0 = ft * 16 + hrow;
    const int f1 = f0 + 16;
#pragma unroll 2
    for (int k0 = 0; k0 < D; k0 += 32) {
      v16bf a0, a1, b0, b1;
      const __bf16* ap = Xs + hrow * D + k0 + aBase;
      loadA(ap, a0);
      loadA(ap + 16 * D, a1);                // rows 16..31
      loadB(W1e + (size_t)f0 * D + k0 + bBase, b0);
      loadB(W1e + (size_t)f1 * D + k0 + bBase, b1);
      acc00 = wmma_bf16(a0, b0, acc00);
      acc10 = wmma_bf16(a1, b0, acc10);      // B reuse across M-tiles
      acc01 = wmma_bf16(a0, b1, acc01);
      acc11 = wmma_bf16(a1, b1, acc11);
    }
#pragma unroll
    for (int j = 0; j < 8; ++j) {
      const int m = crow + j;
      Hs[m * F        + ft * 16       + hrow] = (__bf16)gelu_exact(acc00[j]);
      Hs[(16 + m) * F + ft * 16       + hrow] = (__bf16)gelu_exact(acc10[j]);
      Hs[m * F        + (ft + 1) * 16 + hrow] = (__bf16)gelu_exact(acc01[j]);
      Hs[(16 + m) * F + (ft + 1) * 16 + hrow] = (__bf16)gelu_exact(acc11[j]);
    }
  }
  __syncthreads();

  // per-lane gates/tokens for the 16 C rows this lane owns (8 per M-tile)
  int   myTok[16];
  float myGate[16];
#pragma unroll
  for (int j = 0; j < 8; ++j) {
    const int m0 = crow + j;            // M-tile 0
    const int m1 = 16 + crow + j;       // M-tile 1
    const bool v0 = (n0 + m0) < c;
    const bool v1 = (n0 + m1) < c;
    myTok[j]      = v0 ? tokIdx[e * N + n0 + m0] : 0;
    myGate[j]     = v0 ? tokGate[e * N + n0 + m0] : 0.f;
    myTok[8 + j]  = v1 ? tokIdx[e * N + n0 + m1] : 0;
    myGate[8 + j] = v1 ? tokGate[e * N + n0 + m1] : 0.f;
  }

  // ---- phase 2: Y = Hs @ W2e^T; wave wv owns d-tiles [8wv, 8wv+8)
  for (int dt = wv * 8; dt < wv * 8 + 8; dt += 2) {
    v8f acc00 = {}, acc01 = {}, acc10 = {}, acc11 = {};
    const int d0 = dt * 16 + hrow;
    const int d1 = d0 + 16;
#pragma unroll 2
    for (int k0 = 0; k0 < F; k0 += 32) {
      v16bf a0, a1, b0, b1;
      const __bf16* ap = Hs + hrow * F + k0 + aBase;
      loadA(ap, a0);
      loadA(ap + 16 * F, a1);                // rows 16..31
      loadB(W2e + (size_t)d0 * F + k0 + bBase, b0);
      loadB(W2e + (size_t)d1 * F + k0 + bBase, b1);
      acc00 = wmma_bf16(a0, b0, acc00);
      acc10 = wmma_bf16(a1, b0, acc10);
      acc01 = wmma_bf16(a0, b1, acc01);
      acc11 = wmma_bf16(a1, b1, acc11);
    }
#pragma unroll
    for (int j = 0; j < 8; ++j) {
      const float g0 = myGate[j];
      const float g1 = myGate[8 + j];
      if (g0 != 0.f) {  // padded rows have g==0; real gates strictly positive
        float* o = out + (size_t)myTok[j] * D + dt * 16 + hrow;
        atomicAdd(o,      acc00[j] * g0);
        atomicAdd(o + 16, acc01[j] * g0);
      }
      if (g1 != 0.f) {
        float* o = out + (size_t)myTok[8 + j] * D + dt * 16 + hrow;
        atomicAdd(o,      acc10[j] * g1);
        atomicAdd(o + 16, acc11[j] * g1);
      }
    }
  }
}

// ---------------------------------------------------------------------------
extern "C" void kernel_launch(void* const* d_in, const int* in_sizes, int n_in,
                              void* d_out, int out_size, void* d_ws,
                              size_t ws_size, hipStream_t stream) {
  (void)in_sizes; (void)n_in; (void)out_size;
  const float* x  = (const float*)d_in[0];   // [N, D] fp32
  const float* Wr = (const float*)d_in[1];   // [E, D] fp32
  const float* W1 = (const float*)d_in[2];   // [E, F, D] fp32
  const float* W2 = (const float*)d_in[3];   // [E, D, F] fp32
  float* out = (float*)d_out;                // [N, D] fp32

  // workspace: cnt[E] | tokIdx[E*N] | tokGate[E*N] | (W1 bf16 | W2 bf16)
  char* ws = (char*)d_ws;
  int*   cnt     = (int*)ws;
  int*   tokIdx  = (int*)(ws + 256);
  float* tokGate = (float*)(ws + 256 + (size_t)E * N * sizeof(int));
  const size_t dispatchBytes = 256 + 2 * (size_t)E * N * 4;
  const size_t wElems = (size_t)E * F * D;        // 33.5M per tensor
  const bool pre = ws_size >= dispatchBytes + 2 * wElems * sizeof(__bf16);

  const int n4 = N * D / 4;
  moe_zero_kernel<<<(n4 + 255) / 256, 256, 0, stream>>>(out, cnt, n4);
  moe_router_kernel<<<N / 256, 256, 0, stream>>>(x, Wr, cnt, tokIdx, tokGate);

  const size_t smem = (size_t)TILE * D * 2 + (size_t)TILE * F * 2;  // 320KB
  // grid x = tile (fastest), y = expert: same-expert tiles launch adjacently
  // so each expert's bf16 weights stay hot in the 192MB L2.
  if (pre) {
    __bf16* W1b = (__bf16*)(ws + dispatchBytes);
    __bf16* W2b = W1b + wElems;
    const int n8 = (int)(wElems / 8);
    moe_cvt_kernel<<<(n8 + 255) / 256, 256, 0, stream>>>(W1, W1b, n8);
    moe_cvt_kernel<<<(n8 + 255) / 256, 256, 0, stream>>>(W2, W2b, n8);
    moe_expert_kernel<__bf16><<<dim3(NTILES, E), 256, smem, stream>>>(
        x, W1b, W2b, cnt, tokIdx, tokGate, out);
  } else {
    // fallback: stream fp32 weights and convert in-loop
    moe_expert_kernel<float><<<dim3(NTILES, E), 256, smem, stream>>>(
        x, W1, W2, cnt, tokIdx, tokGate, out);
  }
}
